// DifferentialAttention_14121852469525
// MI455X (gfx1250) — matrix operand order repaired and checked by reading, not compiled, not run on hardware
//
#include <hip/hip_runtime.h>
#include <hip/hip_bf16.h>

#define D_MODEL 2048
#define D_HEAD  128
#define N_HEADS 16
#define BB      2
#define SS      2048
#define QK_N    (2 * D_HEAD * N_HEADS)   // 4096
#define MROWS   (BB * SS)                // 4096

typedef __attribute__((ext_vector_type(8)))  __bf16 v8bf;
typedef __attribute__((ext_vector_type(16))) __bf16 v16bf;
typedef __attribute__((ext_vector_type(8)))  float  v8f;

static __device__ __forceinline__ v16bf cat8(v8bf lo, v8bf hi) {
  return __builtin_shufflevector(lo, hi, 0,1,2,3,4,5,6,7,8,9,10,11,12,13,14,15);
}
static __device__ __forceinline__ v8f wmma_bf16(v16bf a, v16bf b, v8f c) {
  // D = A(16x32 bf16) * B(32x16 bf16) + C(16x16 f32)
  return __builtin_amdgcn_wmma_f32_16x16x32_bf16(false, a, false, b, (short)0, c, false, false);
}

// ---------------------------------------------------------------- converts
__global__ void cvt_f32_bf16(const float* __restrict__ src, __bf16* __restrict__ dst, int n) {
  int i = blockIdx.x * blockDim.x + threadIdx.x;
  int stride = gridDim.x * blockDim.x;
  for (; i < n; i += stride) dst[i] = (__bf16)src[i];
}

// W [K][N] f32  ->  Wt [N][K] bf16   (32x32 tiles through padded LDS)
__global__ __launch_bounds__(256) void cvt_transpose_bf16(
    const float* __restrict__ W, __bf16* __restrict__ Wt, int K, int N) {
  __shared__ __bf16 tile[32][33];
  const int tn = blockIdx.x * 32;   // N tile
  const int tk = blockIdx.y * 32;   // K tile
  const int r = threadIdx.x >> 5, c = threadIdx.x & 31;   // 8 x 32
#pragma unroll
  for (int rr = 0; rr < 4; rr++) {
    int k = r + rr * 8;
    tile[k][c] = (__bf16)W[(size_t)(tk + k) * N + tn + c];
  }
  __syncthreads();
#pragma unroll
  for (int rr = 0; rr < 4; rr++) {
    int n = r + rr * 8;
    Wt[(size_t)(tn + n) * K + tk + c] = tile[c][n];
  }
}

// V (b, t, h*128+d) bf16 -> Vt ((b*16+h)*128+d, t) bf16
__global__ void transpose_v(const __bf16* __restrict__ V, __bf16* __restrict__ Vt) {
  int i = blockIdx.x * blockDim.x + threadIdx.x;
  int stride = gridDim.x * blockDim.x;
  const int total = MROWS * D_MODEL;
  for (; i < total; i += stride) {
    int b  = i / (SS * 2048);
    int r  = i - b * (SS * 2048);
    int t  = r / 2048;
    int hd = r & 2047;
    Vt[((size_t)b * 2048 + hd) * SS + t] = V[i];
  }
}

// ---------------------------------------------------------------- GEMM
// C[M,N] = A[M,K] * B[K,N] with B pre-transposed: Bt[N][K]. A/Bt bf16 row-major.
// 256 threads (8 waves), 128x128 block tile, K-step 32, software-pipelined.
template <bool F32OUT>
__global__ __launch_bounds__(256) void gemm_bf16(
    const __bf16* __restrict__ A, const __bf16* __restrict__ Bt,
    void* __restrict__ Cout, int M, int N, int K) {
  __shared__ __bf16 As[128][32];
  __shared__ __bf16 Bs[128][32];   // Bs[n][k]

  const int tid  = threadIdx.x;
  const int wave = tid >> 5, lane = tid & 31;
  const int lm = lane & 15, lg = lane >> 4;
  const int wm = wave >> 2, wn = wave & 3;          // 2x4 wave grid
  const int m0 = blockIdx.y * 128, n0 = blockIdx.x * 128;

  v8f acc[4][2];
#pragma unroll
  for (int mi = 0; mi < 4; mi++)
#pragma unroll
    for (int ni = 0; ni < 2; ni++) acc[mi][ni] = (v8f){};

  const int lrow = tid >> 1, lk = (tid & 1) * 16;   // 128 rows x 2 16-elem chunks
  const __bf16* ap = A  + (size_t)(m0 + lrow) * K + lk;
  const __bf16* bp = Bt + (size_t)(n0 + lrow) * K + lk;

  // preload tile 0 into registers
  v8bf a0 = *(const v8bf*)ap, a1 = *(const v8bf*)(ap + 8);
  v8bf b0 = *(const v8bf*)bp, b1 = *(const v8bf*)(bp + 8);

  for (int k0 = 0; k0 < K; k0 += 32) {
    *(v8bf*)&As[lrow][lk]     = a0;
    *(v8bf*)&As[lrow][lk + 8] = a1;
    *(v8bf*)&Bs[lrow][lk]     = b0;
    *(v8bf*)&Bs[lrow][lk + 8] = b1;
    __syncthreads();

    if (k0 + 32 < K) {   // preload next tile; overlaps the WMMA block below
      ap += 32; bp += 32;
      a0 = *(const v8bf*)ap; a1 = *(const v8bf*)(ap + 8);
      b0 = *(const v8bf*)bp; b1 = *(const v8bf*)(bp + 8);
      __builtin_prefetch(ap + 32, 0, 1);
      __builtin_prefetch(bp + 32, 0, 1);
    }

    v16bf af[4], bfr[2];
#pragma unroll
    for (int mi = 0; mi < 4; mi++) {
      const __bf16* p = &As[wm * 64 + mi * 16 + lm][lg * 8];
      af[mi] = cat8(*(const v8bf*)p, *(const v8bf*)(p + 16));  // K 0..7|8..15, 16..23|24..31
    }
#pragma unroll
    for (int ni = 0; ni < 2; ni++) {
      const __bf16* p = &Bs[wn * 32 + ni * 16 + lm][lg * 8];
      bfr[ni] = cat8(*(const v8bf*)p, *(const v8bf*)(p + 16));  // K 0..7|8..15, 16..23|24..31
    }
#pragma unroll
    for (int mi = 0; mi < 4; mi++)
#pragma unroll
      for (int ni = 0; ni < 2; ni++)
        acc[mi][ni] = wmma_bf16(af[mi], bfr[ni], acc[mi][ni]);
    __syncthreads();
  }

#pragma unroll
  for (int mi = 0; mi < 4; mi++)
#pragma unroll
    for (int ni = 0; ni < 2; ni++) {
      int r0 = m0 + wm * 64 + mi * 16 + lg * 8;
      int c  = n0 + wn * 32 + ni * 16 + lm;
#pragma unroll
      for (int v = 0; v < 8; v++) {
        float val = acc[mi][ni][v];
        if (F32OUT) ((float*)Cout)[(size_t)(r0 + v) * N + c] = val;
        else        ((__bf16*)Cout)[(size_t)(r0 + v) * N + c] = (__bf16)val;
      }
    }
}

// ---------------------------------------------------------------- attention
// One workgroup per (b, h, 16-row stripe). Full score stripes live in LDS (256 KB).
__global__ __launch_bounds__(256) void attn_kernel(
    const __bf16* __restrict__ Q, const __bf16* __restrict__ Kp,
    const __bf16* __restrict__ Vt, const float* __restrict__ lam,
    float* __restrict__ AO) {
  extern __shared__ char smem[];
  float* sc1 = (float*)smem;          // [16][2048]  score1 -> attn1
  float* sc2 = sc1 + 16 * 2048;       // [16][2048]  score2 -> attn2 -> (bf16 overlay) attn_w

  const int tid = threadIdx.x, wave = tid >> 5, lane = tid & 31;
  const int lm = lane & 15, lg = lane >> 4;
  const int s0 = blockIdx.x * 16, h = blockIdx.y, b = blockIdx.z;
  const float scale = 0.08838834764831845f;  // 1/sqrt(128)

  // ---- Q fragments (A matrices), held in registers by every wave
  v16bf aq[2][4];
  {
    size_t qbase = ((size_t)(b * SS + s0 + lm)) * QK_N + h * 256;
#pragma unroll
    for (int comp = 0; comp < 2; comp++)
#pragma unroll
      for (int c = 0; c < 4; c++) {
        const __bf16* p = Q + qbase + comp * 128 + c * 32 + lg * 8;
        aq[comp][c] = cat8(*(const v8bf*)p, *(const v8bf*)(p + 16));
      }
  }

  // ---- scores: each wave owns a 16-key column tile per 128-col step
  const size_t kbase = ((size_t)(b * SS)) * QK_N + h * 256;
  for (int jt = 0; jt < SS / 128; jt++) {
    const int t0 = jt * 128 + wave * 16;
    v8f a1 = (v8f){}, a2 = (v8f){};
    const size_t kt = kbase + (size_t)(t0 + lm) * QK_N;
#pragma unroll
    for (int c = 0; c < 4; c++) {
      const __bf16* p1 = Kp + kt + c * 32 + lg * 8;          // K1: d contiguous
      v16bf bf1 = cat8(*(const v8bf*)p1, *(const v8bf*)(p1 + 16));
      a1 = wmma_bf16(aq[0][c], bf1, a1);
      const __bf16* p2 = Kp + kt + 128 + c * 32 + lg * 8;    // K2
      v16bf bf2 = cat8(*(const v8bf*)p2, *(const v8bf*)(p2 + 16));
      a2 = wmma_bf16(aq[1][c], bf2, a2);
    }
#pragma unroll
    for (int v = 0; v < 8; v++) {
      int m = lg * 8 + v, col = t0 + lm;
      sc1[m * 2048 + col] = a1[v] * scale;
      sc2[m * 2048 + col] = a2[v] * scale;
    }
  }
  __syncthreads();

  // ---- per-wave softmax pipeline: wave handles rows 2*wave, 2*wave+1
  const float lamh = lam[h];
  for (int rr = 0; rr < 2; rr++) {
    const int r = wave * 2 + rr;
    float* s1 = sc1 + r * 2048;
    float* s2 = sc2 + r * 2048;

    // softmax(score1) in place
    float mx = -1e30f;
    for (int c = lane; c < 2048; c += 32) mx = fmaxf(mx, s1[c]);
    for (int o = 16; o > 0; o >>= 1) mx = fmaxf(mx, __shfl_xor(mx, o, 32));
    float sm = 0.f;
    for (int c = lane; c < 2048; c += 32) { float e = __expf(s1[c] - mx); s1[c] = e; sm += e; }
    for (int o = 16; o > 0; o >>= 1) sm += __shfl_xor(sm, o, 32);
    float inv = 1.f / sm;
    for (int c = lane; c < 2048; c += 32) s1[c] *= inv;

    // softmax(score2) in place
    mx = -1e30f;
    for (int c = lane; c < 2048; c += 32) mx = fmaxf(mx, s2[c]);
    for (int o = 16; o > 0; o >>= 1) mx = fmaxf(mx, __shfl_xor(mx, o, 32));
    sm = 0.f;
    for (int c = lane; c < 2048; c += 32) { float e = __expf(s2[c] - mx); s2[c] = e; sm += e; }
    for (int o = 16; o > 0; o >>= 1) sm += __shfl_xor(sm, o, 32);
    inv = 1.f / sm;
    for (int c = lane; c < 2048; c += 32) s2[c] *= inv;

    // outer softmax of (attn1 - lam*attn2)
    mx = -1e30f;
    for (int c = lane; c < 2048; c += 32) mx = fmaxf(mx, s1[c] - lamh * s2[c]);
    for (int o = 16; o > 0; o >>= 1) mx = fmaxf(mx, __shfl_xor(mx, o, 32));
    sm = 0.f;
    for (int c = lane; c < 2048; c += 32) sm += __expf(s1[c] - lamh * s2[c] - mx);
    for (int o = 16; o > 0; o >>= 1) sm += __shfl_xor(sm, o, 32);
    inv = 1.f / sm;

    // bf16 attn_w overlaid over this row's region of sc2 (reads precede writes
    // within each lockstep wave iteration; rows are wave-private; the clobbered
    // float index c/2 is always behind every remaining read index)
    __bf16* aw = (__bf16*)((char*)sc2 + (size_t)r * 2048 * sizeof(float));
    for (int c = lane; c < 2048; c += 32) {
      float w = __expf(s1[c] - lamh * s2[c] - mx) * inv;
      aw[c] = (__bf16)w;
    }
  }
  __syncthreads();

  // ---- out = attn_w(16x2048) @ V(2048x128): wave computes d-tile [wave*16, +16)
  v8f acc = (v8f){};
  const int d0 = wave * 16;
  const size_t vb = ((size_t)(b * 2048) + h * 128 + d0 + lm) * (size_t)SS;
#pragma unroll 4
  for (int kc = 0; kc < 64; kc++) {
    const __bf16* arow =
        (const __bf16*)((const char*)sc2 + (size_t)lm * 2048 * sizeof(float)) + kc * 32 + lg * 8;
    v16bf a = cat8(*(const v8bf*)arow, *(const v8bf*)(arow + 16));
    const __bf16* vp = Vt + vb + kc * 32 + lg * 8;   // t contiguous
    v16bf bv = cat8(*(const v8bf*)vp, *(const v8bf*)(vp + 16));
    acc = wmma_bf16(a, bv, acc);
  }
#pragma unroll
  for (int v = 0; v < 8; v++) {
    int m = lg * 8 + v;
    AO[((size_t)(b * SS + s0 + m)) * D_MODEL + h * 128 + d0 + lm] = acc[v];
  }
}

// ---------------------------------------------------------------- RMS norm
__global__ __launch_bounds__(256) void rms_kernel(
    const float* __restrict__ AO, const float* __restrict__ gamma,
    const float* __restrict__ lam, __bf16* __restrict__ Y) {
  __shared__ float red[8];
  __shared__ float s_inv;
  const int row = blockIdx.x, tid = threadIdx.x;
  const float* x = AO + (size_t)row * D_MODEL;
  float ss = 0.f;
  for (int c = tid; c < D_MODEL; c += 256) { float v = x[c]; ss += v * v; }
  for (int o = 16; o > 0; o >>= 1) ss += __shfl_xor(ss, o, 32);
  if ((tid & 31) == 0) red[tid >> 5] = ss;
  __syncthreads();
  if (tid == 0) {
    float t = 0.f;
    for (int i = 0; i < 8; i++) t += red[i];
    s_inv = rsqrtf(t / (float)D_MODEL + 1e-6f);
  }
  __syncthreads();
  float lmean = 0.f;
  for (int i = 0; i < N_HEADS; i++) lmean += lam[i];
  lmean *= (1.f / N_HEADS);
  const float coef = s_inv * (1.f - lmean);
  for (int c = tid; c < D_MODEL; c += 256)
    Y[(size_t)row * D_MODEL + c] = (__bf16)(x[c] * gamma[c] * coef);
}

// ---------------------------------------------------------------- launch
extern "C" void kernel_launch(void* const* d_in, const int* in_sizes, int n_in,
                              void* d_out, int out_size, void* d_ws, size_t ws_size,
                              hipStream_t stream) {
  const float* x     = (const float*)d_in[0];
  const float* Wq    = (const float*)d_in[1];
  const float* Wk    = (const float*)d_in[2];
  const float* Wv    = (const float*)d_in[3];
  const float* Wo    = (const float*)d_in[4];
  const float* gamma = (const float*)d_in[5];
  const float* lam   = (const float*)d_in[6];

  char* ws = (char*)d_ws;
  size_t off = 0;
  auto alloc = [&](size_t bytes) -> char* {
    char* p = ws + off;
    off += (bytes + 255) & ~(size_t)255;
    return p;
  };
  __bf16* xb   = (__bf16*)alloc((size_t)MROWS * D_MODEL * 2);
  __bf16* wqT  = (__bf16*)alloc((size_t)D_MODEL * QK_N * 2);     // [4096][2048]
  __bf16* wkT  = (__bf16*)alloc((size_t)D_MODEL * QK_N * 2);     // [4096][2048]
  __bf16* wvT  = (__bf16*)alloc((size_t)D_MODEL * D_MODEL * 2);  // [2048][2048]
  __bf16* woT  = (__bf16*)alloc((size_t)D_MODEL * D_MODEL * 2);  // [2048][2048]
  __bf16* Qb   = (__bf16*)alloc((size_t)MROWS * QK_N * 2);
  __bf16* Kb   = (__bf16*)alloc((size_t)MROWS * QK_N * 2);
  __bf16* Vb   = (__bf16*)alloc((size_t)MROWS * D_MODEL * 2);
  __bf16* Vtb  = (__bf16*)alloc((size_t)MROWS * D_MODEL * 2);
  float*  AO   = (float*)alloc((size_t)MROWS * D_MODEL * 4);
  __bf16* Yb   = (__bf16*)alloc((size_t)MROWS * D_MODEL * 2);

  cvt_f32_bf16<<<2048, 256, 0, stream>>>(x, xb, MROWS * D_MODEL);
  // weights: transpose once so GEMM B-staging is fully contiguous
  cvt_transpose_bf16<<<dim3(QK_N / 32, D_MODEL / 32), 256, 0, stream>>>(Wq, wqT, D_MODEL, QK_N);
  cvt_transpose_bf16<<<dim3(QK_N / 32, D_MODEL / 32), 256, 0, stream>>>(Wk, wkT, D_MODEL, QK_N);
  cvt_transpose_bf16<<<dim3(D_MODEL / 32, D_MODEL / 32), 256, 0, stream>>>(Wv, wvT, D_MODEL, D_MODEL);
  cvt_transpose_bf16<<<dim3(D_MODEL / 32, D_MODEL / 32), 256, 0, stream>>>(Wo, woT, D_MODEL, D_MODEL);

  gemm_bf16<false><<<dim3(QK_N / 128, MROWS / 128), 256, 0, stream>>>(
      xb, wqT, Qb, MROWS, QK_N, D_MODEL);
  gemm_bf16<false><<<dim3(QK_N / 128, MROWS / 128), 256, 0, stream>>>(
      xb, wkT, Kb, MROWS, QK_N, D_MODEL);
  gemm_bf16<false><<<dim3(D_MODEL / 128, MROWS / 128), 256, 0, stream>>>(
      xb, wvT, Vb, MROWS, D_MODEL, D_MODEL);

  transpose_v<<<2048, 256, 0, stream>>>(Vb, Vtb);

  const size_t smem = (size_t)2 * 16 * 2048 * sizeof(float);  // 256 KB of 320 KB LDS
  hipFuncSetAttribute((const void*)attn_kernel,
                      hipFuncAttributeMaxDynamicSharedMemorySize, (int)smem);
  attn_kernel<<<dim3(SS / 16, N_HEADS, BB), 256, smem, stream>>>(Qb, Kb, Vtb, lam, AO);

  rms_kernel<<<MROWS, 256, 0, stream>>>(AO, gamma, lam, Yb);

  gemm_bf16<true><<<dim3(D_MODEL / 128, MROWS / 128), 256, 0, stream>>>(
      Yb, woT, d_out, MROWS, D_MODEL, D_MODEL);
}
